// DeformSPPFBlockGatingUnit_44667659878419
// MI455X (gfx1250) — matrix operand order repaired and checked by reading, not compiled, not run on hardware
//
#include <hip/hip_runtime.h>
#include <hip/hip_bf16.h>

// ---- problem sizes ----
constexpr int kN   = 8;
constexpr int kDim = 128;
constexpr int kH   = 192;
constexpr int kW   = 192;
constexpr int kHW  = kH * kW;     // 36864
constexpr int kD4  = 32;
constexpr int kPC  = 8;

typedef __attribute__((ext_vector_type(16))) __bf16 v16bf;
typedef __attribute__((ext_vector_type(8)))  float  v8f;
typedef __attribute__((ext_vector_type(4)))  unsigned v4u;
typedef __attribute__((ext_vector_type(8)))  unsigned v8u;

// ---- helpers ----
__device__ __forceinline__ unsigned short f2us_bf(float f) {
  unsigned u = __builtin_bit_cast(unsigned, f);
  unsigned r = u + 0x7FFFu + ((u >> 16) & 1u);   // round-to-nearest-even
  return (unsigned short)(r >> 16);
}
__device__ __forceinline__ __bf16 f2bf(float f) {
  unsigned short h = f2us_bf(f);
  return __builtin_bit_cast(__bf16, h);
}
__device__ __forceinline__ __bf16 us2bf(unsigned short h) {
  return __builtin_bit_cast(__bf16, h);
}
// A-fragment (16-bit, 16x32): lane<16 rows M=lane, K = {0..7,16..23}; lane>=16 same M, K+8
__device__ __forceinline__ int kkA(int e, int lane) {
  return (e & 7) + ((e >> 3) << 4) + ((lane >= 16) ? 8 : 0);
}
// B-fragment (16-bit, 32x16): lane<16 cols N=lane, K = {0..7,16..23}; lane>=16 same N, K+8
__device__ __forceinline__ int kkB(int e, int lane) {
  return (e & 7) + ((e >> 3) << 4) + ((lane >= 16) ? 8 : 0);
}
__device__ __forceinline__ int imin(int a, int b) { return a < b ? a : b; }
__device__ __forceinline__ int imax(int a, int b) { return a > b ? a : b; }

// ============================================================
// TDM: load a 2-D bf16 tile (tile_w x nrows, row stride = stride_elems)
// from global into LDS at lds_off. D# built per CDNA5 ISA:
//   group0: count=1 | lds_addr | global_addr[56:0] | type=2
//   group1: data_size=2B, tensor_dim0=rowlen, tensor_dim1=nrows,
//           tile_dim0=tile_w, tile_dim1=nrows, dim0_stride=stride
// Issued by one wave; caller does s_wait_tensorcnt + barrier.
// ============================================================
__device__ __forceinline__ void tdm_load_2d_bf16(const unsigned short* gsrc,
                                                 unsigned lds_off,
                                                 unsigned rowlen_elems,
                                                 unsigned nrows,
                                                 unsigned tile_w,
                                                 unsigned stride_elems) {
  unsigned long long ga = (unsigned long long)(const void*)gsrc;
  v4u g0;
  g0[0] = 1u;                                   // count=1, user mode
  g0[1] = lds_off;                              // lds_addr (bytes)
  g0[2] = (unsigned)ga;                         // global_addr[31:0]
  g0[3] = (unsigned)(ga >> 32) | (2u << 30);    // global_addr[56:32] | type=2
  v8u g1;
  g1[0] = (1u << 16);                           // data_size=1 -> 2 bytes
  g1[1] = (rowlen_elems & 0xFFFFu) << 16;       // tensor_dim0[15:0]
  g1[2] = (rowlen_elems >> 16) | ((nrows & 0xFFFFu) << 16);  // dim0[31:16] | dim1[15:0]
  g1[3] = (nrows >> 16) | (tile_w << 16);       // dim1[31:16] | tile_dim0
  g1[4] = nrows & 0xFFFFu;                      // tile_dim1 (tile_dim2=0)
  g1[5] = stride_elems;                         // tensor_dim0_stride[31:0]
  g1[6] = 0u;                                   // stride0[47:32] | stride1[15:0]
  g1[7] = 0u;                                   // stride1[47:16]
  asm volatile("tensor_load_to_lds %0, %1" :: "s"(g0), "s"(g1) : "memory");
}

// ============================================================
// Weight prep: fp32 -> bf16, GEMM-friendly layouts.
// ============================================================
__global__ void prep_weights(const float* __restrict__ w1, const float* __restrict__ w2,
                             const float* __restrict__ bgw, const float* __restrict__ dcw,
                             unsigned short* __restrict__ w1b, unsigned short* __restrict__ w2b,
                             unsigned short* __restrict__ bgwb, unsigned short* __restrict__ dcwb) {
  int i = blockIdx.x * blockDim.x + threadIdx.x;
  if (i < 36864) {        // conv1: 9*128*32 -> [(tap*128+ci)*32+co]
    int co = i & 31, ci = (i >> 5) & 127, tap = i >> 12;
    w1b[i] = f2us_bf(w1[(co * 128 + ci) * 9 + tap]);
  }
  if (i < 147456) {       // conv2: 9*128*128 -> [(tap*128+ci)*128+co]
    int co = i & 127, ci = (i >> 7) & 127, tap = i >> 14;
    w2b[i] = f2us_bf(w2[(co * 128 + ci) * 9 + tap]);
  }
  if (i < 16384) {        // bgu: 4 stages * [k*64+q]
    int q = i & 63, k = (i >> 6) & 63, s = i >> 12;
    bgwb[i] = f2us_bf(bgw[(s * 64 + q) * 64 + k]);
  }
  if (i < 27648) {        // deform: 3 stages * [(tap*32+ci)*32+co]
    int co = i & 31, ci = (i >> 5) & 31, tap = (i >> 10) % 9, s = i / 9216;
    dcwb[i] = f2us_bf(dcw[((s * 32 + co) * 32 + ci) * 9 + tap]);
  }
}

// ============================================================
// Implicit-GEMM 3x3 conv, pad 1, bf16 WMMA, f32 accumulate.
// Block = 8 waves sharing one cout-tile. The full B tile
// (CIN*9 rows x 16 couts, bf16) is TDM-loaded into LDS once, then
// the K-loop runs barrier-free: A gathered from global, B from LDS.
// ============================================================
template <int CIN, int COUT>
__global__ void conv3x3_wmma(const float* __restrict__ src,
                             const unsigned short* __restrict__ wbf,
                             const float* __restrict__ bias,
                             float* __restrict__ dst) {
  constexpr int nChunks = (CIN * 9) / 32;
  __shared__ unsigned short lb[nChunks * 32 * 16];   // [K][16] tile
  const int lane = threadIdx.x & 31;
  const int wave = threadIdx.x >> 5;
  constexpr int nCT = COUT / 16;
  const int coutTile = blockIdx.x % nCT;
  const int pixGroup = blockIdx.x / nCT;
  const int pixTile  = pixGroup * 8 + wave;
  constexpr int tilesPerRow = kW / 16;
  const int n   = pixTile / (kH * tilesPerRow);
  const int rem = pixTile % (kH * tilesPerRow);
  const int y   = rem / tilesPerRow;
  const int x0  = (rem % tilesPerRow) * 16;
  const int m   = lane & 15;
  const int coutBase = coutTile * 16;

  if (wave == 0) {
    tdm_load_2d_bf16(wbf + coutBase, (unsigned)(unsigned long long)(void*)lb,
                     COUT, CIN * 9, 16, COUT);
    __builtin_amdgcn_s_wait_tensorcnt(0);
  }
  __syncthreads();

  v8f c = {};
  for (int ch = 0; ch < nChunks; ++ch) {
    const int tap  = (ch * 32) / CIN;
    const int cin0 = (ch * 32) % CIN;
    const int dy = tap / 3 - 1, dx = tap % 3 - 1;

    const int  yy    = y + dy;
    const bool rowok = (unsigned)yy < (unsigned)kH;
    const int  xx    = x0 + m + dx;
    const bool colok = (unsigned)xx < (unsigned)kW;
    const float* srow = src + ((n * CIN) * kH + yy) * kW;   // + cin*kHW per element
    if (rowok && colok) __builtin_prefetch(srow + cin0 * kHW + xx, 0, 0);

    v16bf a, b;
#pragma unroll
    for (int e = 0; e < 16; ++e) {
      const int cin = cin0 + kkA(e, lane);
      float v = 0.f;
      if (rowok && colok) v = srow[cin * kHW + xx];
      a[e] = f2bf(v);
      b[e] = us2bf(lb[(ch * 32 + kkB(e, lane)) * 16 + (lane & 15)]);
    }
    c = __builtin_amdgcn_wmma_f32_16x16x32_bf16(false, a, false, b, (short)0, c, false, false);
  }

  const float bb   = bias[coutBase + (lane & 15)];
  const int   mofs = (lane >= 16) ? 8 : 0;
  float* drow = dst + ((n * COUT + coutBase + (lane & 15)) * kH + y) * kW + x0;
#pragma unroll
  for (int r = 0; r < 8; ++r) drow[mofs + r] = c[r] + bb;
}

// ============================================================
// Block gating unit as GEMM; 64x64 bf16 B tile TDM-loaded to LDS.
// ============================================================
__global__ void bgu_wmma(const float* __restrict__ src,
                         const unsigned short* __restrict__ wbf,  // [64][64] k-major
                         const float* __restrict__ bias,          // [64]
                         float* __restrict__ cat, int catOfs) {
  __shared__ unsigned short lgw[64 * 64];
  const int lane = threadIdx.x & 31;
  const int wave = threadIdx.x >> 5;
  const int wg    = blockIdx.x * 8 + wave;
  const int ntile = wg & 3;
  const int mtile = wg >> 2;
  const int m      = lane & 15;
  const int block0 = mtile * 16;

  if (wave == 0) {
    tdm_load_2d_bf16(wbf, (unsigned)(unsigned long long)(void*)lgw, 64, 64, 64, 64);
    __builtin_amdgcn_s_wait_tensorcnt(0);
  }
  __syncthreads();

  v8f c = {};
#pragma unroll
  for (int chk = 0; chk < 2; ++chk) {
    const int blk  = block0 + m;
    const int n    = blk / (kD4 * 576);
    const int r2   = blk % (kD4 * 576);
    const int cch  = r2 / 576;
    const int bidx = r2 % 576;
    const int by = bidx / 24, bx = bidx % 24;
    const float* sbase = src + ((n * kD4 + cch) * kH + by * 8) * kW + bx * 8;

    v16bf a, b;
#pragma unroll
    for (int e = 0; e < 16; ++e) {
      const int k  = chk * 32 + kkA(e, lane);
      const int ky = k >> 3, kx = k & 7;
      a[e] = f2bf(sbase[ky * kW + kx]);
      b[e] = us2bf(lgw[(chk * 32 + kkB(e, lane)) * 64 + ntile * 16 + (lane & 15)]);
    }
    c = __builtin_amdgcn_wmma_f32_16x16x32_bf16(false, a, false, b, (short)0, c, false, false);
  }

  const int   q  = ntile * 16 + (lane & 15);
  const float bq = bias[q];
  const int p8 = q >> 3, q8 = q & 7;
  const int mofs = (lane >= 16) ? 8 : 0;
#pragma unroll
  for (int r = 0; r < 8; ++r) {
    const int blk  = block0 + mofs + r;
    const int n    = blk / (kD4 * 576);
    const int r2   = blk % (kD4 * 576);
    const int cch  = r2 / 576;
    const int bidx = r2 % 576;
    const int by = bidx / 24, bx = bidx % 24;
    const int yy = by * 8 + p8, xx = bx * 8 + q8;
    const float g = c[r] + bq;
    cat[((n * kDim + catOfs + cch) * kH + yy) * kW + xx] =
        src[((n * kD4 + cch) * kH + yy) * kW + xx] * g;
  }
}

// ============================================================
// Fused pconv (two 3x3 convs on first 8 channels) + 1x1 offset/mask
// heads + sigmoid. One thread per pixel; weights cached in LDS.
// ============================================================
__global__ void pconv_offmask(const float* __restrict__ src,
                              const float* __restrict__ off_pw,  // 8*8*9
                              const float* __restrict__ off_w,   // 18*32
                              const float* __restrict__ off_b,   // 18
                              const float* __restrict__ mask_pw, // 8*8*9
                              const float* __restrict__ mask_w,  // 9*32
                              float* __restrict__ off, float* __restrict__ msk) {
  __shared__ float s_opw[576], s_mpw[576], s_ow[576], s_mw[288], s_ob[18];
  for (int t = threadIdx.x; t < 576; t += blockDim.x) {
    s_opw[t] = off_pw[t]; s_mpw[t] = mask_pw[t]; s_ow[t] = off_w[t];
  }
  for (int t = threadIdx.x; t < 288; t += blockDim.x) s_mw[t] = mask_w[t];
  if (threadIdx.x < 18) s_ob[threadIdx.x] = off_b[threadIdx.x];
  __syncthreads();

  const int p = blockIdx.x * blockDim.x + threadIdx.x;   // over kN*kHW, exact
  const int n = p / kHW, pix = p % kHW;
  const int y = pix / kW, x = pix % kW;

  float po[kPC], pm[kPC];
#pragma unroll
  for (int i = 0; i < kPC; ++i) { po[i] = 0.f; pm[i] = 0.f; }

  for (int ci = 0; ci < kPC; ++ci) {
    const float* sc = src + (n * kD4 + ci) * kHW;
#pragma unroll
    for (int ky = 0; ky < 3; ++ky) {
      const int yy = y + ky - 1;
      if ((unsigned)yy >= (unsigned)kH) continue;
#pragma unroll
      for (int kx = 0; kx < 3; ++kx) {
        const int xx = x + kx - 1;
        if ((unsigned)xx >= (unsigned)kW) continue;
        const float v = sc[yy * kW + xx];
        const int wo = ci * 9 + ky * 3 + kx;
#pragma unroll
        for (int o = 0; o < kPC; ++o) {
          po[o] += s_opw[o * 72 + wo] * v;
          pm[o] += s_mpw[o * 72 + wo] * v;
        }
      }
    }
  }
  float xin[kD4];
  for (int i = kPC; i < kD4; ++i) xin[i] = src[(n * kD4 + i) * kHW + y * kW + x];

#pragma unroll
  for (int j = 0; j < 18; ++j) {
    float acc = s_ob[j];
    for (int i = 0; i < kPC; ++i)   acc += s_ow[j * 32 + i] * po[i];
    for (int i = kPC; i < kD4; ++i) acc += s_ow[j * 32 + i] * xin[i];
    off[(n * 18 + j) * kHW + y * kW + x] = acc;
  }
#pragma unroll
  for (int t = 0; t < 9; ++t) {
    float acc = 0.f;                 // mask head has no bias
    for (int i = 0; i < kPC; ++i)   acc += s_mw[t * 32 + i] * pm[i];
    for (int i = kPC; i < kD4; ++i) acc += s_mw[t * 32 + i] * xin[i];
    msk[(n * 9 + t) * kHW + y * kW + x] = 1.f / (1.f + __expf(-acc));
  }
}

// ============================================================
// Deformable 3x3 conv as gathered implicit GEMM.
// 288x32 bf16 weight tile TDM-loaded to LDS once per block.
// ============================================================
__global__ void deform_wmma(const float* __restrict__ src,
                            const float* __restrict__ off,
                            const float* __restrict__ msk,
                            const unsigned short* __restrict__ wdc,  // [288][32]
                            float* __restrict__ dst) {
  __shared__ unsigned short ldc[288 * 32];
  const int lane = threadIdx.x & 31;
  const int wave = threadIdx.x >> 5;
  const int pixTile = blockIdx.x * 8 + wave;
  constexpr int tilesPerRow = kW / 16;
  const int n   = pixTile / (kH * tilesPerRow);
  const int rem = pixTile % (kH * tilesPerRow);
  const int y   = rem / tilesPerRow;
  const int x0  = (rem % tilesPerRow) * 16;
  const int m = lane & 15;
  const int x = x0 + m;

  if (wave == 0) {
    tdm_load_2d_bf16(wdc, (unsigned)(unsigned long long)(void*)ldc, 32, 288, 32, 32);
    __builtin_amdgcn_s_wait_tensorcnt(0);
  }
  __syncthreads();

  v8f c0 = {}, c1 = {};
  for (int t = 0; t < 9; ++t) {
    const int dy = t / 3 - 1, dx = t % 3 - 1;
    const float oy = off[((n * 18 + 2 * t) * kH + y) * kW + x];
    const float ox = off[((n * 18 + 2 * t + 1) * kH + y) * kW + x];
    const float mv = msk[((n * 9 + t) * kH + y) * kW + x];
    float sy = (float)(y + dy) + oy;
    float sx = (float)(x + dx) + ox;
    sy = fminf(fmaxf(sy, -2.f), (float)(kH + 1));
    sx = fminf(fmaxf(sx, -2.f), (float)(kW + 1));
    const float fy = floorf(sy), fx = floorf(sx);
    const int y0 = (int)fy, xi0 = (int)fx;
    const float wy = sy - fy, wx = sx - fx;
    const float w00 = (1.f - wy) * (1.f - wx) * mv, w01 = (1.f - wy) * wx * mv;
    const float w10 = wy * (1.f - wx) * mv,         w11 = wy * wx * mv;
    const bool v00 = ((unsigned)y0 < (unsigned)kH) & ((unsigned)xi0 < (unsigned)kW);
    const bool v01 = ((unsigned)y0 < (unsigned)kH) & ((unsigned)(xi0 + 1) < (unsigned)kW);
    const bool v10 = ((unsigned)(y0 + 1) < (unsigned)kH) & ((unsigned)xi0 < (unsigned)kW);
    const bool v11 = ((unsigned)(y0 + 1) < (unsigned)kH) & ((unsigned)(xi0 + 1) < (unsigned)kW);
    const int yc0 = imin(imax(y0, 0), kH - 1), yc1 = imin(imax(y0 + 1, 0), kH - 1);
    const int xc0 = imin(imax(xi0, 0), kW - 1), xc1 = imin(imax(xi0 + 1, 0), kW - 1);
    const float* sb = src + n * kD4 * kHW;

    v16bf a, b0, b1;
#pragma unroll
    for (int e = 0; e < 16; ++e) {
      const int cin = kkA(e, lane);
      const float* sc = sb + cin * kHW;
      const float a00 = v00 ? sc[yc0 * kW + xc0] : 0.f;
      const float a01 = v01 ? sc[yc0 * kW + xc1] : 0.f;
      const float a10 = v10 ? sc[yc1 * kW + xc0] : 0.f;
      const float a11 = v11 ? sc[yc1 * kW + xc1] : 0.f;
      a[e] = f2bf(a00 * w00 + a01 * w01 + a10 * w10 + a11 * w11);
      const int kb = t * 32 + kkB(e, lane);
      b0[e] = us2bf(ldc[kb * 32 + (lane & 15)]);
      b1[e] = us2bf(ldc[kb * 32 + 16 + (lane & 15)]);
    }
    c0 = __builtin_amdgcn_wmma_f32_16x16x32_bf16(false, a, false, b0, (short)0, c0, false, false);
    c1 = __builtin_amdgcn_wmma_f32_16x16x32_bf16(false, a, false, b1, (short)0, c1, false, false);
  }

  const int mofs = (lane >= 16) ? 8 : 0;
  const int co0  = lane & 15;
#pragma unroll
  for (int r = 0; r < 8; ++r) {
    const int xx = x0 + mofs + r;
    dst[((n * kD4 + co0) * kH + y) * kW + xx]      = fmaxf(c0[r], 0.f);
    dst[((n * kD4 + 16 + co0) * kH + y) * kW + xx] = fmaxf(c1[r], 0.f);
  }
}

// ============================================================
extern "C" void kernel_launch(void* const* d_in, const int* in_sizes, int n_in,
                              void* d_out, int out_size, void* d_ws, size_t ws_size,
                              hipStream_t stream) {
  (void)in_sizes; (void)n_in; (void)out_size; (void)ws_size;
  const float* x     = (const float*)d_in[0];
  const float* w1    = (const float*)d_in[1];
  const float* b1    = (const float*)d_in[2];
  const float* bgw   = (const float*)d_in[3];
  const float* bgb   = (const float*)d_in[4];
  const float* dcw   = (const float*)d_in[5];
  const float* offpw = (const float*)d_in[6];
  const float* offw  = (const float*)d_in[7];
  const float* offb  = (const float*)d_in[8];
  const float* mskpw = (const float*)d_in[9];
  const float* mskw  = (const float*)d_in[10];
  const float* w2    = (const float*)d_in[11];
  const float* b2    = (const float*)d_in[12];
  float* out = (float*)d_out;

  char* ws = (char*)d_ws;
  size_t ofs = 0;
  auto alloc = [&](size_t bytes) {
    void* p = ws + ofs;
    ofs = (ofs + bytes + 255) & ~(size_t)255;
    return p;
  };
  float* xa  = (float*)alloc((size_t)kN * kD4 * kHW * 4);
  float* xb  = (float*)alloc((size_t)kN * kD4 * kHW * 4);
  float* cat = (float*)alloc((size_t)kN * kDim * kHW * 4);
  float* off = (float*)alloc((size_t)kN * 18 * kHW * 4);
  float* msk = (float*)alloc((size_t)kN * 9 * kHW * 4);
  unsigned short* w1b  = (unsigned short*)alloc(36864 * 2);
  unsigned short* w2b  = (unsigned short*)alloc(147456 * 2);
  unsigned short* bgwb = (unsigned short*)alloc(16384 * 2);
  unsigned short* dcwb = (unsigned short*)alloc(27648 * 2);

  prep_weights<<<576, 256, 0, stream>>>(w1, w2, bgw, dcw, w1b, w2b, bgwb, dcwb);

  // conv1: 128 -> 32  (pixTiles=18432 -> 2304 groups x 2 cout tiles)
  conv3x3_wmma<128, 32><<<4608, 256, 0, stream>>>(x, w1b, b1, xa);
  // BGU stage 0 -> cat channels [0,32)
  bgu_wmma<<<4608, 256, 0, stream>>>(xa, bgwb, bgb, cat, 0);

  float* cur = xa;
  float* nxt = xb;
  for (int s = 0; s < 3; ++s) {
    pconv_offmask<<<1152, 256, 0, stream>>>(cur, offpw + s * 576, offw + s * 576,
                                            offb + s * 18, mskpw + s * 576,
                                            mskw + s * 288, off, msk);
    deform_wmma<<<2304, 256, 0, stream>>>(cur, off, msk, dcwb + s * 9216, nxt);
    bgu_wmma<<<4608, 256, 0, stream>>>(nxt, bgwb + (s + 1) * 4096, bgb + (s + 1) * 64,
                                       cat, (s + 1) * kD4);
    float* tmp = cur; cur = nxt; nxt = tmp;
  }
  // final conv: 128 -> 128 from concat (2304 groups x 8 cout tiles)
  conv3x3_wmma<128, 128><<<18432, 256, 0, stream>>>(cat, w2b, b2, out);
}
